// CorrLayer_19172734009699
// MI455X (gfx1250) — matrix operand order repaired and checked by reading, not compiled, not run on hardware
//
#include <hip/hip_runtime.h>
#include <hip/hip_bf16.h>
#include <math.h>

typedef __attribute__((ext_vector_type(16))) _Float16 v16h;
typedef __attribute__((ext_vector_type(8)))  _Float16 h8;
typedef __attribute__((ext_vector_type(4)))  _Float16 h4;
typedef __attribute__((ext_vector_type(8)))  float    v8f;

typedef union { v16h v; h8 h[2]; } v16h_u;

// Problem constants
#define BB   32
#define LL   1536
#define DD   512
#define HH   8
#define CH   16384      // 4 * 8 * 512 channels
#define NF   769        // rfft bins for L=1536
#define F2   1600       // padded (800 re | 800 im)
#define TOPK 14
#define NROW 49152      // B*L

// ---------------------------------------------------------------------------
// f16-WMMA GEMM, fully specialized via template:  C = A @ B (+bias)
// 128x64 tile per 256-thread (8-wave) workgroup, 2x2 WMMA tiles per wave,
// K-step 32, double-buffered LDS (global loads overlap WMMA).
// A staged k-major; B staged TRANSPOSED (n-major) so fragments are
// contiguous 32B per lane -> ds_load_b128 pairs for both operands.
// MODE 0: plain f32 row-major store
// MODE 1: scatter into [channel,time] layout, f16 (projection Q/K epilogue)
// MODE 2: scatter into [channel,time] layout, f32 (projection V epilogue)
// ---------------------------------------------------------------------------
template <int AH, int BH, int MODE>
__global__ __launch_bounds__(256) void wmma_gemm(
    const void* __restrict__ Aptr,
    const void* __restrict__ Bptr,
    const float* __restrict__ bias,
    void* __restrict__ Cptr,
    int M, int N, int K)
{
  __shared__ _Float16 As[2][128][40];   // row stride 80B (16B-aligned)
  __shared__ _Float16 BsT[2][64][40];   // transposed: [n][k], stride 80B

  const int tid  = threadIdx.x;
  const int lane = tid & 31;
  const int wave = tid >> 5;
  const int wm   = wave >> 1;           // 0..3 (M direction, 32 rows each)
  const int wn   = wave & 1;            // 0..1 (N direction, 32 cols each)
  const int m0   = blockIdx.y * 128;
  const int n0   = blockIdx.x * 64;

  const float*    Af = (const float*)Aptr;
  const _Float16* Ah = (const _Float16*)Aptr;
  const float*    Bf = (const float*)Bptr;
  const _Float16* Bh = (const _Float16*)Bptr;

  // Staging registers (next tile in flight while WMMA works on current)
  h8     aH[2];  float4 aF[4];
  h8     bH;     float4 bF4[2];

  auto load_tile = [&](int k0) {
    if (AH) {
      #pragma unroll
      for (int i = 0; i < 2; ++i) {
        int idx = (tid + i * 256) * 8;          // 4096 halves
        int r = idx >> 5, c = idx & 31;
        aH[i] = *(const h8*)(Ah + (size_t)(m0 + r) * K + (size_t)(k0 + c));
      }
    } else {
      #pragma unroll
      for (int i = 0; i < 4; ++i) {
        int idx = (tid + i * 256) * 4;          // 4096 floats
        int r = idx >> 5, c = idx & 31;
        aF[i] = *(const float4*)(Af + (size_t)(m0 + r) * K + (size_t)(k0 + c));
      }
    }
    if (BH) {
      int idx = tid * 8;                        // 2048 halves
      int r = idx >> 6, c = idx & 63;
      bH = *(const h8*)(Bh + (size_t)(k0 + r) * N + (size_t)(n0 + c));
    } else {
      #pragma unroll
      for (int i = 0; i < 2; ++i) {
        int idx = (tid + i * 256) * 4;
        int r = idx >> 6, c = idx & 63;
        bF4[i] = *(const float4*)(Bf + (size_t)(k0 + r) * N + (size_t)(n0 + c));
      }
    }
  };

  auto store_tile = [&](int buf) {
    if (AH) {
      #pragma unroll
      for (int i = 0; i < 2; ++i) {
        int idx = (tid + i * 256) * 8;
        int r = idx >> 5, c = idx & 31;
        *(h8*)&As[buf][r][c] = aH[i];
      }
    } else {
      #pragma unroll
      for (int i = 0; i < 4; ++i) {
        int idx = (tid + i * 256) * 4;
        int r = idx >> 5, c = idx & 31;
        h4 h;
        h[0] = (_Float16)aF[i].x; h[1] = (_Float16)aF[i].y;
        h[2] = (_Float16)aF[i].z; h[3] = (_Float16)aF[i].w;
        *(h4*)&As[buf][r][c] = h;
      }
    }
    if (BH) {
      int idx = tid * 8;
      int r = idx >> 6, c = idx & 63;
      #pragma unroll
      for (int e = 0; e < 8; ++e) BsT[buf][c + e][r] = bH[e];
    } else {
      #pragma unroll
      for (int i = 0; i < 2; ++i) {
        int idx = (tid + i * 256) * 4;
        int r = idx >> 6, c = idx & 63;
        BsT[buf][c + 0][r] = (_Float16)bF4[i].x;
        BsT[buf][c + 1][r] = (_Float16)bF4[i].y;
        BsT[buf][c + 2][r] = (_Float16)bF4[i].z;
        BsT[buf][c + 3][r] = (_Float16)bF4[i].w;
      }
    }
  };

  v8f acc[2][2];
  #pragma unroll
  for (int r = 0; r < 2; ++r)
    #pragma unroll
    for (int s = 0; s < 2; ++s)
      #pragma unroll
      for (int e = 0; e < 8; ++e) acc[r][s][e] = 0.0f;

  // Prologue: fill buffer 0
  load_tile(0);
  store_tile(0);
  __syncthreads();

  int cur = 0;
  for (int k0 = 0; k0 < K; k0 += 32) {
    const bool has_next = (k0 + 32 < K);
    if (has_next) load_tile(k0 + 32);   // HBM loads in flight during WMMA

    // ---- Fragments (contiguous k = 16*(lane/16) + s, same for A and B) ----
    const int kb = (lane >> 4) << 3;    // (unused) 0 or 8
    const int kh = (lane >> 4) << 4;    // B: 0 or 16
    v16h afrag[2], bfrag[2];
    #pragma unroll
    for (int r = 0; r < 2; ++r) {
      int row = wm * 32 + r * 16 + (lane & 15);
      v16h_u u;
      u.h[0] = *(const h8*)&As[cur][row][kh];        // K = kh..kh+7
      u.h[1] = *(const h8*)&As[cur][row][kh + 8];    // K = kh+8..kh+15
      afrag[r] = u.v;
    }
    #pragma unroll
    for (int s = 0; s < 2; ++s) {
      int col = wn * 32 + s * 16 + (lane & 15);
      bfrag[s] = *(const v16h*)&BsT[cur][col][kh];   // K = kh..kh+15 contiguous
    }
    #pragma unroll
    for (int r = 0; r < 2; ++r)
      #pragma unroll
      for (int s = 0; s < 2; ++s)
        acc[r][s] = __builtin_amdgcn_wmma_f32_16x16x32_f16(
            false, afrag[r], false, bfrag[s], (short)0, acc[r][s], false, false);

    if (has_next) {
      store_tile(cur ^ 1);
      __syncthreads();
      cur ^= 1;
    }
  }

  // ---- Epilogue ----
  #pragma unroll
  for (int r = 0; r < 2; ++r)
    #pragma unroll
    for (int s = 0; s < 2; ++s)
      #pragma unroll
      for (int i = 0; i < 8; ++i) {
        int m = m0 + wm * 32 + r * 16 + i + ((lane >> 4) << 3);
        int n = n0 + wn * 32 + s * 16 + (lane & 15);
        float v = acc[r][s][i];
        if (bias) v += bias[n];
        if (MODE == 0) {
          ((float*)Cptr)[(size_t)m * N + n] = v;
        } else {
          // m = b*1536 + l (projection row), n = feature c
          int b  = m / LL, l = m - b * LL;
          int bp = b >> 3,  mm = b & 7;
          int hp = l & 7,   j  = l >> 3;
          int t  = mm * 192 + j;
          size_t dst = (size_t)(((bp << 3) + hp) * DD + n) * LL + t;
          if (MODE == 1) ((_Float16*)Cptr)[dst] = (_Float16)v;
          else           ((float*)Cptr)[dst]    = v;
        }
      }
}

// ---------------------------------------------------------------------------
// DFT basis fills (constant matrices, built on device)
// ---------------------------------------------------------------------------
__global__ void fill_fwd_basis(_Float16* __restrict__ bF) {
  int idx = blockIdx.x * blockDim.x + threadIdx.x;     // [LL x F2]
  if (idx >= LL * F2) return;
  int t = idx / F2, f = idx - t * F2;
  int fi = (f < 800) ? f : f - 800;
  float v = 0.0f;
  if (fi < NF) {
    int p = (int)(((long long)t * fi) % LL);
    float ang = 6.283185307179586f * (float)p / (float)LL;
    v = (f < 800) ? __cosf(ang) : -__sinf(ang);
  }
  bF[idx] = (_Float16)v;
}

__global__ void fill_inv_basis(_Float16* __restrict__ bI) {
  int idx = blockIdx.x * blockDim.x + threadIdx.x;     // [F2 x LL]
  if (idx >= F2 * LL) return;
  int k = idx / LL, tau = idx - k * LL;
  int fi = (k < 800) ? k : k - 800;
  float v = 0.0f;
  if (fi < NF) {
    float alpha = (fi == 0 || fi == 768) ? 1.0f : 2.0f;
    int p = (int)(((long long)fi * tau) % LL);
    float ang = 6.283185307179586f * (float)p / (float)LL;
    float base = (k < 800) ? __cosf(ang) : -__sinf(ang);
    v = alpha * base * (1.0f / (float)LL);
  }
  bI[idx] = (_Float16)v;
}

// W = Qf * conj(Kf), elementwise over spectra, f16 output for inverse GEMM
__global__ void spectral_product(const float* __restrict__ Qf,
                                 const float* __restrict__ Kf,
                                 _Float16* __restrict__ Wf) {
  int idx = blockIdx.x * blockDim.x + threadIdx.x;     // CH * 800
  if (idx >= CH * 800) return;
  int ch = idx / 800, f = idx - ch * 800;
  size_t base = (size_t)ch * F2;
  float qr = Qf[base + f], qi = Qf[base + 800 + f];
  float kr = Kf[base + f], ki = Kf[base + 800 + f];
  Wf[base + f]       = (_Float16)(qr * kr + qi * ki);
  Wf[base + 800 + f] = (_Float16)(qi * kr - qr * ki);
}

// ---------------------------------------------------------------------------
// Top-14 + softmax: one wave32 per channel, corr row staged in LDS
// ---------------------------------------------------------------------------
__global__ __launch_bounds__(256) void topk_kernel(
    const float* __restrict__ corr,
    int* __restrict__ delays, float* __restrict__ wts) {
  __shared__ float buf[8][LL];
  int wave = threadIdx.x >> 5, lane = threadIdx.x & 31;
  int ch = blockIdx.x * 8 + wave;
  float* row = buf[wave];
  const float* src = corr + (size_t)ch * LL;
  for (int i = lane; i < LL; i += 32) row[i] = src[i];

  float vals[TOPK]; int idxs[TOPK];
  for (int k = 0; k < TOPK; ++k) {
    float best = -1.0e30f; int bi = 0;
    for (int i = lane; i < LL; i += 32) {
      float v = row[i];
      if (v > best) { best = v; bi = i; }
    }
    for (int off = 16; off > 0; off >>= 1) {
      float ov = __shfl_down(best, off, 32);
      int   oi = __shfl_down(bi,   off, 32);
      if (ov > best) { best = ov; bi = oi; }
    }
    best = __shfl(best, 0, 32);
    bi   = __shfl(bi,   0, 32);
    vals[k] = best; idxs[k] = bi;
    if (lane == 0) row[bi] = -1.0e30f;
  }
  if (lane == 0) {
    float mx = vals[0], s = 0.0f, e[TOPK];
    for (int k = 0; k < TOPK; ++k) { e[k] = __expf(vals[k] - mx); s += e[k]; }
    float inv = 1.0f / s;
    for (int k = 0; k < TOPK; ++k) {
      wts[(size_t)ch * TOPK + k]    = e[k] * inv;
      delays[(size_t)ch * TOPK + k] = idxs[k];
    }
  }
}

// ---------------------------------------------------------------------------
// Delay aggregation: one workgroup per channel; writes f16 rows for final GEMM
// ---------------------------------------------------------------------------
__global__ __launch_bounds__(256) void agg_kernel(
    const float* __restrict__ Vc,
    const int* __restrict__ delays, const float* __restrict__ wts,
    _Float16* __restrict__ G) {
  __shared__ float vrow[LL];
  __shared__ float w[TOPK];
  __shared__ int   d[TOPK];
  int ch = blockIdx.x, tid = threadIdx.x;
  for (int i = tid; i < LL; i += 256) vrow[i] = Vc[(size_t)ch * LL + i];
  if (tid < TOPK) {
    w[tid] = wts[(size_t)ch * TOPK + tid];
    d[tid] = delays[(size_t)ch * TOPK + tid];
  }
  __syncthreads();
  int c  = ch & (DD - 1);
  int hp = (ch >> 9) & 7;
  int bp = ch >> 12;
  for (int t = tid; t < LL; t += 256) {
    float acc = 0.0f;
    #pragma unroll
    for (int k = 0; k < TOPK; ++k) {
      int tt = t + d[k];
      if (tt >= LL) tt -= LL;
      acc += w[k] * vrow[tt];
    }
    int m = t / 192, j = t - m * 192;
    int row = (bp * 8 + m) * LL + hp + 8 * j;
    G[(size_t)row * DD + c] = (_Float16)acc;
  }
}

// ---------------------------------------------------------------------------
extern "C" void kernel_launch(void* const* d_in, const int* in_sizes, int n_in,
                              void* d_out, int out_size, void* d_ws, size_t ws_size,
                              hipStream_t stream) {
  (void)in_sizes; (void)n_in; (void)out_size; (void)ws_size;
  const float* queries = (const float*)d_in[0];
  const float* keys    = (const float*)d_in[1];
  const float* values  = (const float*)d_in[2];
  const float* Wq = (const float*)d_in[3];
  const float* bq = (const float*)d_in[4];
  const float* Wk = (const float*)d_in[5];
  const float* bk = (const float*)d_in[6];
  const float* Wv = (const float*)d_in[7];
  const float* bv = (const float*)d_in[8];
  const float* Wo = (const float*)d_in[9];
  const float* bo = (const float*)d_in[10];

  char* ws = (char*)d_ws;
  size_t off = 0;
  auto alloc = [&](size_t bytes) -> void* {
    void* p = (void*)(ws + off);
    off += (bytes + 255) & ~(size_t)255;
    return p;
  };

  _Float16* Qc = (_Float16*)alloc((size_t)CH * LL * 2);
  _Float16* Kc = (_Float16*)alloc((size_t)CH * LL * 2);
  float*    Vc = (float*)   alloc((size_t)CH * LL * 4);
  float*    Qf = (float*)   alloc((size_t)CH * F2 * 4);
  float*    Kf = (float*)   alloc((size_t)CH * F2 * 4);
  _Float16* Wf = (_Float16*)alloc((size_t)CH * F2 * 2);
  _Float16* bF = (_Float16*)alloc((size_t)LL * F2 * 2);
  _Float16* bI = (_Float16*)alloc((size_t)F2 * LL * 2);
  int*   delays = (int*)  alloc((size_t)CH * TOPK * 4);
  float* wts    = (float*)alloc((size_t)CH * TOPK * 4);
  // Dead-region aliases (Qf/Kf are consumed by spectral_product)
  float*    corr = Qf;               // CH*LL*4 <= CH*F2*4
  _Float16* G    = (_Float16*)Kf;    // NROW*DD*2 <= CH*F2*4

  fill_fwd_basis<<<(LL * F2 + 255) / 256, 256, 0, stream>>>(bF);
  fill_inv_basis<<<(F2 * LL + 255) / 256, 256, 0, stream>>>(bI);

  // Projections: scatter into [channel,time] layout (Q/K f16, V f32)
  wmma_gemm<0, 0, 1><<<dim3(DD / 64, NROW / 128), 256, 0, stream>>>(
      queries, Wq, bq, Qc, NROW, DD, DD);
  wmma_gemm<0, 0, 1><<<dim3(DD / 64, NROW / 128), 256, 0, stream>>>(
      keys, Wk, bk, Kc, NROW, DD, DD);
  wmma_gemm<0, 0, 2><<<dim3(DD / 64, NROW / 128), 256, 0, stream>>>(
      values, Wv, bv, Vc, NROW, DD, DD);

  // Forward DFT as GEMM against constant basis
  wmma_gemm<1, 1, 0><<<dim3(F2 / 64, CH / 128), 256, 0, stream>>>(
      Qc, bF, nullptr, Qf, CH, F2, LL);
  wmma_gemm<1, 1, 0><<<dim3(F2 / 64, CH / 128), 256, 0, stream>>>(
      Kc, bF, nullptr, Kf, CH, F2, LL);

  spectral_product<<<(CH * 800 + 255) / 256, 256, 0, stream>>>(Qf, Kf, Wf);

  // Inverse DFT (Hermitian weights folded into basis) -> corr
  wmma_gemm<1, 1, 0><<<dim3(LL / 64, CH / 128), 256, 0, stream>>>(
      Wf, bI, nullptr, corr, CH, LL, F2);

  topk_kernel<<<CH / 8, 256, 0, stream>>>(corr, delays, wts);
  agg_kernel<<<CH, 256, 0, stream>>>(Vc, delays, wts, G);

  // Output projection -> d_out (f32)
  wmma_gemm<1, 0, 0><<<dim3(DD / 64, NROW / 128), 256, 0, stream>>>(
      G, Wo, bo, (float*)d_out, NROW, DD, DD);
}
